// PointerWithLatentRelations_22170621182363
// MI455X (gfx1250) — matrix operand order repaired and checked by reading, not compiled, not run on hardware
//
#include <hip/hip_runtime.h>

#define HDIM 512
#define LDIM 1024
#define CDIM 512
#define NUM_REL 51
#define NPAD 64

typedef __bf16 bf16;
typedef bf16  v16bf __attribute__((ext_vector_type(16)));
typedef bf16  v8bf  __attribute__((ext_vector_type(8)));
typedef float v8f   __attribute__((ext_vector_type(8)));

__device__ __forceinline__ v8bf ld8(const bf16* p) {
  return *reinterpret_cast<const v8bf*>(p);  // 16B -> global_load_b128
}
__device__ __forceinline__ v16bf cat16(v8bf lo, v8bf hi) {
  return __builtin_shufflevector(lo, hi, 0,1,2,3,4,5,6,7,8,9,10,11,12,13,14,15);
}
__device__ __forceinline__ v8f wmma_bf(v16bf a, v16bf b, v8f c) {
  return __builtin_amdgcn_wmma_f32_16x16x32_bf16(false, a, false, b, (short)0, c,
                                                 false, false);
}

// ---------- bf16 fragment loaders (contiguous, guard-free) ----------
// A 16x32: element e<8 -> K = 8*half + e ; e>=8 -> K = 16 + 8*half + (e-8)
__device__ __forceinline__ v16bf load_a_bf(const bf16* __restrict__ A, int lda,
                                           int m0, int kbase) {
  const int lane = threadIdx.x & 31;
  const bf16* p = A + (size_t)(m0 + (lane & 15)) * lda + kbase + (lane >> 4) * 8;
  return cat16(ld8(p), ld8(p + 16));
}
// B 32x16 from row-major B[n][k]: element e<8 -> K = 8*half + e ; e>=8 -> K = 16 + 8*half + (e-8)
__device__ __forceinline__ v16bf load_b_bf(const bf16* __restrict__ B, int ldb,
                                           int n0, int kbase) {
  const int lane = threadIdx.x & 31;
  const bf16* p = B + (size_t)(n0 + (lane & 15)) * ldb + kbase + (lane >> 4) * 8;
  return cat16(ld8(p), ld8(p + 16));
}

// ---------- f32 fragment loaders (projection kernels; convert once) ----------
__device__ __forceinline__ v16bf load_a_f32(const float* __restrict__ A, int lda,
                                            int m0, int kbase) {
  const int lane = threadIdx.x & 31;
  const float* p = A + (size_t)(m0 + (lane & 15)) * lda + kbase + (lane >> 4) * 8;
  v16bf a;
#pragma unroll
  for (int i = 0; i < 8; ++i) a[i] = (bf16)p[i];
#pragma unroll
  for (int i = 0; i < 8; ++i) a[8 + i] = (bf16)p[16 + i];
  return a;
}
__device__ __forceinline__ v16bf load_b_f32(const float* __restrict__ B, int ldb,
                                            int n0, int kbase) {
  const int lane = threadIdx.x & 31;
  const float* p = B + (size_t)(n0 + (lane & 15)) * ldb + kbase + (lane >> 4) * 8;
  v16bf b;
#pragma unroll
  for (int i = 0; i < 16; ++i) b[i] = (bf16)p[i + (i >> 3) * 8];
  return b;
}

// Projection: D(bf16) = A(f32, MxK) @ B(f32, NxK)^T + bias.
// 128 threads = 4 waves; each wave owns a 16x64 strip (4 accumulators).
__global__ void __launch_bounds__(128) proj_gemm_bf16(
    const float* __restrict__ A, const float* __restrict__ B,
    const float* __restrict__ bias, bf16* __restrict__ Cout,
    int N, int K, int lda, int ldb, int ldc) {
  const int m0 = blockIdx.y * 16;
  const int n0 = (blockIdx.x * 4 + (threadIdx.x >> 5)) * 64;
  if (n0 >= N) return;  // wave-uniform
  v8f acc[4] = {};
  for (int k = 0; k < K; k += 32) {
    v16bf a = load_a_f32(A, lda, m0, k);
#pragma unroll
    for (int t = 0; t < 4; ++t) {
      v16bf b = load_b_f32(B, ldb, n0 + 16 * t, k);
      acc[t] = wmma_bf(a, b, acc[t]);
    }
  }
  const int lane = threadIdx.x & 31;
  const int half = lane >> 4;
#pragma unroll
  for (int t = 0; t < 4; ++t) {
    const int col = n0 + 16 * t + (lane & 15);
    const float bb = bias[col];
#pragma unroll
    for (int r = 0; r < 8; ++r)  // C/D layout: VGPR r -> M = r + 8*half
      Cout[(size_t)(m0 + r + half * 8) * ldc + col] = (bf16)(acc[t][r] + bb);
  }
}

// D(f32) = A(bf16, MxK) @ B(bf16, NxK)^T. NT tiles of 16 cols per wave.
template <int NT>
__global__ void __launch_bounds__(128) gemm_bf16_f32(
    const bf16* __restrict__ A, const bf16* __restrict__ B,
    float* __restrict__ Cout, int N, int K, int lda, int ldb, int ldc) {
  const int m0 = blockIdx.y * 16;
  const int n0 = (blockIdx.x * 4 + (threadIdx.x >> 5)) * (16 * NT);
  if (n0 >= N) return;  // wave-uniform
  v8f acc[NT] = {};
  for (int k = 0; k < K; k += 32) {
    if (k + 64 < K)  // global_prefetch_b8 next-next A strip
      __builtin_prefetch(A + (size_t)(m0 + (threadIdx.x & 15)) * lda + k + 64, 0, 3);
    v16bf a = load_a_bf(A, lda, m0, k);
    v16bf b[NT];
#pragma unroll
    for (int t = 0; t < NT; ++t) b[t] = load_b_bf(B, ldb, n0 + 16 * t, k);
#pragma unroll
    for (int t = 0; t < NT; ++t) acc[t] = wmma_bf(a, b[t], acc[t]);
  }
  const int lane = threadIdx.x & 31;
  const int half = lane >> 4;
#pragma unroll
  for (int t = 0; t < NT; ++t) {
    const int col = n0 + 16 * t + (lane & 15);
#pragma unroll
    for (int r = 0; r < 8; ++r)
      Cout[(size_t)(m0 + r + half * 8) * ldc + col] = acc[t][r];
  }
}

// rel_k_emb (f32, 51x512) -> bf16 64x512, rows 51..63 zero (kills GEMM guards).
__global__ void __launch_bounds__(256) pad_convert_relk(
    const float* __restrict__ in, bf16* __restrict__ out) {
  const int idx = blockIdx.x * 256 + threadIdx.x;  // over NPAD*HDIM
  const int r = idx / HDIM;
  out[idx] = (r < NUM_REL) ? (bf16)in[idx] : (bf16)0.0f;
}

// scores[l,c] = (qk[l,c] + qrel[l, rel[l, qlen+c]]) / sqrt(H); row softmax.
__global__ void __launch_bounds__(256) gather_softmax(
    const float* __restrict__ qk, const float* __restrict__ qrel,
    const int* __restrict__ rel, const int* __restrict__ qlen_ptr,
    float* __restrict__ out) {
  const int l = blockIdx.x;
  const int tid = threadIdx.x;
  const int qlen = *qlen_ptr;
  const float scale = 0.04419417382415922f;  // 1/sqrt(512)
  __shared__ float red[8];

  const int c0 = tid, c1 = tid + 256;
  const int i0 = rel[(size_t)l * LDIM + qlen + c0];
  const int i1 = rel[(size_t)l * LDIM + qlen + c1];
  float s0 = (qk[(size_t)l * CDIM + c0] + qrel[(size_t)l * NPAD + i0]) * scale;
  float s1 = (qk[(size_t)l * CDIM + c1] + qrel[(size_t)l * NPAD + i1]) * scale;

  float m = fmaxf(s0, s1);
#pragma unroll
  for (int off = 16; off > 0; off >>= 1) m = fmaxf(m, __shfl_xor(m, off, 32));
  if ((tid & 31) == 0) red[tid >> 5] = m;
  __syncthreads();
  float bm = red[0];
#pragma unroll
  for (int i = 1; i < 8; ++i) bm = fmaxf(bm, red[i]);
  __syncthreads();

  s0 = __expf(s0 - bm);
  s1 = __expf(s1 - bm);
  float sum = s0 + s1;
#pragma unroll
  for (int off = 16; off > 0; off >>= 1) sum += __shfl_xor(sum, off, 32);
  if ((tid & 31) == 0) red[tid >> 5] = sum;
  __syncthreads();
  float tot = 0.0f;
#pragma unroll
  for (int i = 0; i < 8; ++i) tot += red[i];
  const float inv = 1.0f / tot;

  out[(size_t)l * CDIM + c0] = s0 * inv;
  out[(size_t)l * CDIM + c1] = s1 * inv;
}

extern "C" void kernel_launch(void* const* d_in, const int* in_sizes, int n_in,
                              void* d_out, int out_size, void* d_ws, size_t ws_size,
                              hipStream_t stream) {
  (void)in_sizes; (void)n_in; (void)out_size; (void)ws_size;
  const float* query = (const float*)d_in[0];
  const float* key   = (const float*)d_in[1];
  // d_in[2] value, d_in[5] c_len, d_in[10..11] Wv/bv, d_in[13] rel_v_emb: dead
  const int*   rel   = (const int*)d_in[3];
  const int*   qlen  = (const int*)d_in[4];
  const float* Wq    = (const float*)d_in[6];
  const float* bq    = (const float*)d_in[7];
  const float* Wk    = (const float*)d_in[8];
  const float* bk    = (const float*)d_in[9];
  const float* rel_k = (const float*)d_in[12];
  float* out = (float*)d_out;

  // workspace layout (f32 first, then bf16; all offsets 32B-aligned)
  float* qrel_f  = (float*)d_ws;                    // LDIM*NPAD  f32 (256 KB)
  float* qk_f    = qrel_f + (size_t)LDIM * NPAD;    // LDIM*CDIM  f32 (2 MB)
  bf16*  q_bf    = (bf16*)(qk_f + (size_t)LDIM * CDIM);  // LDIM*HDIM bf16 (1 MB)
  bf16*  k_bf    = q_bf + (size_t)LDIM * HDIM;      // CDIM*HDIM bf16 (512 KB)
  bf16*  relk_bf = k_bf + (size_t)CDIM * HDIM;      // NPAD*HDIM bf16 (64 KB)

  const dim3 blk(128);
  // rel_k_emb -> bf16, zero-padded to 64 rows
  pad_convert_relk<<<dim3(NPAD * HDIM / 256), dim3(256), 0, stream>>>(rel_k, relk_bf);
  // q = bf16(query @ Wq^T + bq)
  proj_gemm_bf16<<<dim3(HDIM / 256, LDIM / 16), blk, 0, stream>>>(
      query, Wq, bq, q_bf, HDIM, HDIM, HDIM, HDIM, HDIM);
  // k = bf16(key @ Wk^T + bk)
  proj_gemm_bf16<<<dim3(HDIM / 256, CDIM / 16), blk, 0, stream>>>(
      key, Wk, bk, k_bf, HDIM, HDIM, HDIM, HDIM, HDIM);
  // qrel_all = q @ relk_pad^T  (64 cols, one 16-col tile per wave)
  gemm_bf16_f32<1><<<dim3(1, LDIM / 16), blk, 0, stream>>>(
      q_bf, relk_bf, qrel_f, NPAD, HDIM, HDIM, HDIM, NPAD);
  // qk = q @ k^T  (4 x 16-col tiles per wave)
  gemm_bf16_f32<4><<<dim3(CDIM / 256, LDIM / 16), blk, 0, stream>>>(
      q_bf, k_bf, qk_f, CDIM, HDIM, HDIM, HDIM, CDIM);
  // fused gather + scale + softmax
  gather_softmax<<<dim3(LDIM), dim3(256), 0, stream>>>(qk_f, qrel_f, rel, qlen, out);
}
